// QuantMatMul_19078244729412
// MI455X (gfx1250) — 8-bit matrix operand order repaired and checked by reading, compiled, not run on hardware
//
#include <hip/hip_runtime.h>
#include <stdint.h>

typedef __attribute__((ext_vector_type(16))) int   v16i;
typedef __attribute__((ext_vector_type(8)))  float v8f;
typedef __attribute__((ext_vector_type(2)))  int   v2i;
typedef __attribute__((ext_vector_type(4)))  int   v4i;

#define BATCH 8
#define MDIM  4096
#define KDIM  512
#define NDIM  4096

#if __has_builtin(__builtin_amdgcn_global_load_async_to_lds_b128)
#define HAVE_ASYNC_LDS 1
#else
#define HAVE_ASYNC_LDS 0
#endif

__device__ __forceinline__ void wait_async0() {
#if HAVE_ASYNC_LDS
#if __has_builtin(__builtin_amdgcn_s_wait_asynccnt)
    __builtin_amdgcn_s_wait_asynccnt(0);
#else
    asm volatile("s_wait_asynccnt 0x0" ::: "memory");
#endif
#endif
}

__device__ __forceinline__ unsigned char f32_to_e4m3(float x) {
    unsigned u = __float_as_uint(x);
    unsigned s = (u >> 24) & 0x80u;
    float ax = fabsf(x);
    if (ax < 0.015625f) {
        int m = (int)rintf(ax * 512.0f);
        return (unsigned char)(s | (unsigned)m);
    }
    if (ax > 464.0f) return (unsigned char)(s | 0x7Eu);
    unsigned ur = u + 0x7FFFFu + ((u >> 20) & 1u);
    int e = (int)((ur >> 23) & 0xFFu) - 127 + 7;
    unsigned m3 = (ur >> 20) & 0x7u;
    if (e > 15 || (e == 15 && m3 == 7u)) return (unsigned char)(s | 0x7Eu);
    return (unsigned char)(s | ((unsigned)e << 3) | m3);
}

__device__ __forceinline__ float scale_from_bits(unsigned bits) {
    return fmaxf(__uint_as_float(bits) * (1.0f / 448.0f), 1e-12f);
}

__global__ void qmm_zero_kernel(unsigned* p) {
    if (threadIdx.x < 2) p[threadIdx.x] = 0u;
}

__global__ void qmm_amax_kernel(const float4* __restrict__ x, int n4, unsigned* out) {
    __shared__ float red[256];
    int stride = gridDim.x * blockDim.x;
    float m = 0.0f;
    for (int i = blockIdx.x * blockDim.x + threadIdx.x; i < n4; i += stride) {
        float4 v = x[i];
        m = fmaxf(m, fmaxf(fmaxf(fabsf(v.x), fabsf(v.y)), fmaxf(fabsf(v.z), fabsf(v.w))));
    }
    red[threadIdx.x] = m;
    __syncthreads();
    for (int off = 128; off > 0; off >>= 1) {
        if ((int)threadIdx.x < off)
            red[threadIdx.x] = fmaxf(red[threadIdx.x], red[threadIdx.x + off]);
        __syncthreads();
    }
    if (threadIdx.x == 0) atomicMax(out, __float_as_uint(red[0]));
}

__global__ void qmm_quantA_kernel(const float4* __restrict__ x, unsigned* __restrict__ q,
                                  const unsigned* __restrict__ amax, int n4) {
    float inv = 1.0f / scale_from_bits(*amax);
    int i = blockIdx.x * 256 + threadIdx.x;
    if (i < n4) {
        float4 v = x[i];
        unsigned b0 = f32_to_e4m3(v.x * inv);
        unsigned b1 = f32_to_e4m3(v.y * inv);
        unsigned b2 = f32_to_e4m3(v.z * inv);
        unsigned b3 = f32_to_e4m3(v.w * inv);
        q[i] = b0 | (b1 << 8) | (b2 << 16) | (b3 << 24);
    }
}

__global__ void qmm_quantTB_kernel(const float* __restrict__ x, unsigned char* __restrict__ q,
                                   const unsigned* __restrict__ amax) {
    __shared__ unsigned char lT[64 * 72];
    float inv = 1.0f / scale_from_bits(*amax);
    int b  = blockIdx.z;
    int k0 = blockIdx.y * 64;
    int n0 = blockIdx.x * 64;
    const float* src = x + (size_t)b * KDIM * NDIM;
    int t = threadIdx.x;
#pragma unroll
    for (int i = 0; i < 16; ++i) {
        int lin = i * 256 + t;
        int r = lin >> 6, c = lin & 63;
        float v = src[(size_t)(k0 + r) * NDIM + (n0 + c)];
        lT[c * 72 + r] = f32_to_e4m3(v * inv);
    }
    __syncthreads();
    int nl = t >> 2, kb = (t & 3) * 16;
    unsigned w[4];
#pragma unroll
    for (int j = 0; j < 4; ++j) {
        const unsigned char* p = &lT[nl * 72 + kb + j * 4];
        w[j] = (unsigned)p[0] | ((unsigned)p[1] << 8) | ((unsigned)p[2] << 16) | ((unsigned)p[3] << 24);
    }
    *(uint4*)(q + (size_t)((size_t)b * NDIM + n0 + nl) * KDIM + k0 + kb) =
        make_uint4(w[0], w[1], w[2], w[3]);
}

__device__ __forceinline__ void stage_tile(int* dstA, int* dstB,
                                           const uint4* Ag, const uint4* Bg,
                                           int kt, int tid) {
#pragma unroll
    for (int i = 0; i < 4; ++i) {
        int id = tid + 256 * i;
        int r = id >> 3, c4 = id & 7;
        size_t gidx = (size_t)r * (KDIM / 16) + (size_t)kt * 8 + c4;
        int lidx = r * 8 + c4;
#if HAVE_ASYNC_LDS
        __builtin_amdgcn_global_load_async_to_lds_b128(
            (v4i*)(Ag + gidx), (v4i*)(((uint4*)dstA) + lidx), 0, 0);
        __builtin_amdgcn_global_load_async_to_lds_b128(
            (v4i*)(Bg + gidx), (v4i*)(((uint4*)dstB) + lidx), 0, 0);
#else
        ((uint4*)dstA)[lidx] = Ag[gidx];
        ((uint4*)dstB)[lidx] = Bg[gidx];
#endif
    }
}

__global__ __launch_bounds__(256, 2)
void qmm_gemm_kernel(const unsigned char* __restrict__ Aq, const unsigned char* __restrict__ Bq,
                     float* __restrict__ out, const unsigned* __restrict__ amax) {
    __shared__ int lA[2][128 * 32];
    __shared__ int lB[2][128 * 32];

    float sc = scale_from_bits(amax[0]) * scale_from_bits(amax[1]);

    int b  = blockIdx.z;
    int m0 = blockIdx.y * 128;
    int n0 = blockIdx.x * 128;
    int tid  = threadIdx.x;
    int lane = tid & 31;
    int wid  = tid >> 5;
    int wm = (wid & 1) * 64;
    int wn = (wid >> 1) * 32;

    v8f acc[4][2];
#pragma unroll
    for (int mi = 0; mi < 4; ++mi)
#pragma unroll
        for (int ni = 0; ni < 2; ++ni)
            acc[mi][ni] = (v8f)(0.0f);

    const uint4* Ag = (const uint4*)(Aq + (size_t)((size_t)b * MDIM + m0) * KDIM);
    const uint4* Bg = (const uint4*)(Bq + (size_t)((size_t)b * NDIM + n0) * KDIM);

    int kh = lane >> 4;
    int ll = lane & 15;

    stage_tile(lA[0], lB[0], Ag, Bg, 0, tid);

    const int KT = KDIM / 128;
    for (int kt = 0; kt < KT; ++kt) {
        int cur = kt & 1;
        wait_async0();
        __syncthreads();

        if (kt + 1 < KT)
            stage_tile(lA[1 - cur], lB[1 - cur], Ag, Bg, kt + 1, tid);

        v16i afr[4];
#pragma unroll
        for (int mi = 0; mi < 4; ++mi) {
            int row = wm + mi * 16 + ll;
            const int* base = lA[cur] + row * 32 + kh * 4;
            v4i* dst = (v4i*)&afr[mi];
#pragma unroll
            for (int j = 0; j < 4; ++j) dst[j] = *(const v4i*)(base + j * 8);
        }
        v16i bfr[2];
#pragma unroll
        for (int ni = 0; ni < 2; ++ni) {
            int col = wn + ni * 16 + ll;
            const int* base = lB[cur] + col * 32 + kh * 4;
            v4i* dst = (v4i*)&bfr[ni];
#pragma unroll
            for (int qd = 0; qd < 4; ++qd) dst[qd] = *(const v4i*)(base + qd * 8);
        }

#pragma unroll
        for (int mi = 0; mi < 4; ++mi)
#pragma unroll
            for (int ni = 0; ni < 2; ++ni)
                acc[mi][ni] = __builtin_amdgcn_wmma_f32_16x16x128_fp8_fp8(
                    afr[mi], bfr[ni], (short)0, acc[mi][ni], false, false);
    }

    float* obase = out + (size_t)b * MDIM * NDIM;
#pragma unroll
    for (int mi = 0; mi < 4; ++mi) {
#pragma unroll
        for (int ni = 0; ni < 2; ++ni) {
            int row0 = m0 + wm + mi * 16 + kh * 8;
            int col  = n0 + wn + ni * 16 + ll;
            float* o = obase + (size_t)row0 * NDIM + col;
#pragma unroll
            for (int j = 0; j < 8; ++j)
                o[(size_t)j * NDIM] = acc[mi][ni][j] * sc;
        }
    }
}

extern "C" void kernel_launch(void* const* d_in, const int* in_sizes, int n_in,
                              void* d_out, int out_size, void* d_ws, size_t ws_size,
                              hipStream_t stream) {
    const float* in1 = (const float*)d_in[0];
    const float* in2 = (const float*)d_in[1];
    float* out = (float*)d_out;

    unsigned* amax = (unsigned*)d_ws;
    unsigned char* Aq = (unsigned char*)d_ws + 256;
    unsigned char* Bq = Aq + (size_t)BATCH * MDIM * KDIM;

    const int n_elems = BATCH * MDIM * KDIM;
    const int n4 = n_elems / 4;

    qmm_zero_kernel<<<1, 32, 0, stream>>>(amax);
    qmm_amax_kernel<<<4096, 256, 0, stream>>>((const float4*)in1, n4, amax + 0);
    qmm_amax_kernel<<<4096, 256, 0, stream>>>((const float4*)in2, n4, amax + 1);
    qmm_quantA_kernel<<<(n4 + 255) / 256, 256, 0, stream>>>((const float4*)in1, (unsigned*)Aq, amax + 0, n4);
    qmm_quantTB_kernel<<<dim3(NDIM / 64, KDIM / 64, BATCH), 256, 0, stream>>>(in2, Bq, amax + 1);
    qmm_gemm_kernel<<<dim3(NDIM / 128, MDIM / 128, BATCH), 256, 0, stream>>>(Aq, Bq, out, amax);
}
